// Mamba3D_66571993088874
// MI455X (gfx1250) — matrix operand order repaired and checked by reading, not compiled, not run on hardware
//
#include <hip/hip_runtime.h>

// ---------------------------------------------------------------------------
// Types (avoid HIP's non-trivial vector structs; use clang ext vectors)
// ---------------------------------------------------------------------------
typedef __attribute__((ext_vector_type(16))) __bf16 bf16x16;
typedef __attribute__((ext_vector_type(8)))  __bf16 bf16x8;
typedef __attribute__((ext_vector_type(8)))  float  f32x8;
typedef __attribute__((ext_vector_type(4)))  unsigned int u32x4;
typedef unsigned short ushort_t;

#define NC 20
#define NH 11
#define NW 11
#define NPOS (NC*NH*NW)   // 2420
#define DD 128
#define BB 128
#define PD 90

__device__ __forceinline__ ushort_t f2bf(float f) {
    unsigned u = __float_as_uint(f);
    unsigned r = u + 0x7FFFu + ((u >> 16) & 1u);   // round-to-nearest-even
    return (ushort_t)(r >> 16);
}
__device__ __forceinline__ float bf2f(ushort_t h) {
    return __uint_as_float(((unsigned)h) << 16);
}

// Block reduction over blockDim.x == 128 threads
__device__ __forceinline__ float block_sum(float v, float* sbuf) {
    int t = threadIdx.x;
    sbuf[t] = v; __syncthreads();
    #pragma unroll
    for (int s = 64; s > 0; s >>= 1) {
        if (t < s) sbuf[t] += sbuf[t + s];
        __syncthreads();
    }
    float r = sbuf[0]; __syncthreads();
    return r;
}

// ---------------------------------------------------------------------------
// K0: transpose weights to Wt[n][k] bf16 : [W_in^T | W1^T | W2^T | W3^T]
// ---------------------------------------------------------------------------
__global__ void k_prep_w(const float* __restrict__ W_in,
                         const float* __restrict__ W_h,
                         ushort_t* __restrict__ Wt) {
    int n = blockIdx.x;     // 0..127 output dim
    int k = threadIdx.x;    // 0..127 input dim
    Wt[0*16384 + n*128 + k] = f2bf(W_in[k*128 + n]);
    Wt[1*16384 + n*128 + k] = f2bf(W_h[(      k)*128 + n]);
    Wt[2*16384 + n*128 + k] = f2bf(W_h[(128 + k)*128 + n]);
    Wt[3*16384 + n*128 + k] = f2bf(W_h[(256 + k)*128 + n]);
}

// ---------------------------------------------------------------------------
// K1: patchify + LN(90) + embed (90->128) + LN(128) + sincos3D posemb
//     -> xe[pos][b][d] bf16   (one block per position, loop over batch)
// ---------------------------------------------------------------------------
__global__ __launch_bounds__(128) void k_embed(
    const float* __restrict__ hsi,
    const float* __restrict__ lnp_g, const float* __restrict__ lnp_b,
    const float* __restrict__ W_emb, const float* __restrict__ b_emb,
    const float* __restrict__ lne_g, const float* __restrict__ lne_b,
    ushort_t* __restrict__ xe) {
    int pos = blockIdx.x;                // (ci*11 + hi)*11 + wi
    int wi = pos % NW, hi = (pos / NW) % NH, ci = pos / (NH * NW);
    int t = threadIdx.x;
    __shared__ float sv[96];
    __shared__ float red[128];

    // positional embedding for channel t (computed once)
    float pe = 0.f;
    if (t < 126) {
        int grp = t / 21, idx = t % 21;
        float om = powf(10000.f, -(float)idx / 20.f);
        float coord = (grp < 2) ? (float)wi : (grp < 4) ? (float)hi : (float)ci;
        float ang = coord * om;
        pe = (grp & 1) ? cosf(ang) : sinf(ang);
    }

    for (int b = 0; b < BB; b++) {
        float v = 0.f;
        if (t < PD) {
            int p1 = t / 30, rem = t % 30, p2 = rem / 10, pc = rem % 10;
            int c = ci * 10 + pc, hh = hi * 3 + p1, ww = wi * 3 + p2;
            v = hsi[((size_t)b * 200 + c) * 1089 + hh * 33 + ww];
        }
        float m  = block_sum(t < PD ? v : 0.f, red) / (float)PD;
        float dv = (t < PD) ? (v - m) : 0.f;
        float var = block_sum(dv * dv, red) / (float)PD;
        float nv = (t < PD) ? (dv * rsqrtf(var + 1e-5f) * lnp_g[t] + lnp_b[t]) : 0.f;
        if (t < 96) sv[t] = (t < PD) ? nv : 0.f;
        __syncthreads();

        float e = b_emb[t];
        for (int k = 0; k < PD; k++) e += sv[k] * W_emb[k * DD + t];

        float m2 = block_sum(e, red) / (float)DD;
        float d2 = e - m2;
        float v2 = block_sum(d2 * d2, red) / (float)DD;
        float y  = d2 * rsqrtf(v2 + 1e-5f) * lne_g[t] + lne_b[t] + pe;
        xe[((size_t)pos * BB + b) * DD + t] = f2bf(y);
        __syncthreads();
    }
}

// ---------------------------------------------------------------------------
// K2: one 3D-scan wavefront step.
//  h[pos] = tanh(src0[pos]@W_in + h[i-1]@W1 + h[j-1]@W2 + h[k-1]@W3 + b_in+b_h)
//  Block = 256 thr = 8 waves; wave w owns batch tile M=16w, all 8 N tiles.
//  Weights staged via LDS (32KB, one source matrix at a time).
// ---------------------------------------------------------------------------
__global__ __launch_bounds__(256) void k_scan(
    const ushort_t* __restrict__ src0,   // xe (layer1) or h1 (layer2)
    ushort_t* __restrict__ hbuf,         // current layer hidden state
    const ushort_t* __restrict__ Wt,     // 4 x [128][128] bf16, transposed
    const float* __restrict__ b_in, const float* __restrict__ b_h,
    int d) {
    __shared__ __align__(32) ushort_t sW[16384];   // 32 KB: one weight matrix

    int i = blockIdx.x / NH, j = blockIdx.x % NH;  // grid = 220
    int k = d - i - j;
    if (k < 0 || k >= NW) return;                  // uniform exit
    int pos = (i * NH + j) * NW + k;

    int lane = threadIdx.x & 31;
    int w    = threadIdx.x >> 5;                   // M tile (batch rows 16w..)

    const ushort_t* Abase[4];
    Abase[0] = src0 + (size_t)pos * (BB * DD);
    Abase[1] = (i > 0) ? hbuf + (size_t)(pos - NH * NW) * (BB * DD) : nullptr;
    Abase[2] = (j > 0) ? hbuf + (size_t)(pos - NW) * (BB * DD) : nullptr;
    Abase[3] = (k > 0) ? hbuf + (size_t)(pos - 1) * (BB * DD) : nullptr;

    f32x8 acc[8] = {};

    // Operand k order used by this code (wave32, 16-bit; A and B match):
    //  A 16x32: lane l -> row M=l&15, K block (l>>4)*16 .. +15 (contiguous)
    //  B 32x16: lane l -> col N=l&15, K block (l>>4)*16 .. +15 (contiguous)
    int arow  = (w << 4) + (lane & 15);
    int akoff = (lane >> 4) << 4;     // 0 or 16
    int bcol  = lane & 15;
    int bkoff = (lane >> 4) << 4;     // 0 or 16

    for (int s = 0; s < 4; s++) {
        __syncthreads();
        {   // cooperative stage of Wt[s] into LDS: 2048 x 16B / 256 thr
            const u32x4* g  = (const u32x4*)(Wt + s * 16384);
            u32x4*       sh = (u32x4*)sW;
            #pragma unroll
            for (int it = 0; it < 8; it++)
                sh[threadIdx.x + it * 256] = g[threadIdx.x + it * 256];
        }
        __syncthreads();
        const ushort_t* A = Abase[s];
        if (!A) continue;                          // uniform (zero init state)
        #pragma unroll
        for (int kk = 0; kk < 4; kk++) {
            const ushort_t* ap = A + (size_t)arow * DD + (kk << 5) + akoff;
            bf16x8 alo = *(const bf16x8*)(ap);
            bf16x8 ahi = *(const bf16x8*)(ap + 8);
            bf16x16 af = __builtin_shufflevector(alo, ahi,
                0, 1, 2, 3, 4, 5, 6, 7, 8, 9, 10, 11, 12, 13, 14, 15);
            #pragma unroll
            for (int n = 0; n < 8; n++) {
                const ushort_t* bp = sW + ((n << 4) + bcol) * DD + (kk << 5) + bkoff;
                bf16x16 bfm = *(const bf16x16*)bp;
                acc[n] = __builtin_amdgcn_wmma_f32_16x16x32_bf16(
                    false, af, false, bfm, (short)0, acc[n], false, false);
            }
        }
    }

    // epilogue: bias + tanh -> bf16 store (C/D layout: lane l, VGPR v ->
    //  M = (l>>4)*8 + v, N = l&15 within the tile)
    int mbase = (w << 4) + ((lane >> 4) << 3);
    #pragma unroll
    for (int n = 0; n < 8; n++) {
        int col = (n << 4) + bcol;
        float bias = b_in[col] + b_h[col];
        #pragma unroll
        for (int v = 0; v < 8; v++) {
            float f = tanhf(acc[n][v] + bias);
            hbuf[(size_t)pos * (BB * DD) + (size_t)(mbase + v) * DD + col] = f2bf(f);
        }
    }
}

// ---------------------------------------------------------------------------
// K3: final LN+tanh at last position, head LN, MLP 128->128->16
//     grid=128 (one block per batch row), block=128
// ---------------------------------------------------------------------------
__global__ __launch_bounds__(128) void k_head(
    const ushort_t* __restrict__ h2,
    const float* __restrict__ g_out, const float* __restrict__ b_out,
    const float* __restrict__ g_hd,  const float* __restrict__ b_hd,
    const float* __restrict__ W1h,   const float* __restrict__ b1h,
    const float* __restrict__ W2h,   const float* __restrict__ b2h,
    float* __restrict__ out) {
    int b = blockIdx.x, t = threadIdx.x;
    __shared__ float red[128];
    __shared__ float sz[128];
    __shared__ float st[128];

    const size_t posL = (size_t)(NPOS - 1);
    float x = bf2f(h2[(posL * BB + b) * DD + t]);

    float m  = block_sum(x, red) / (float)DD;
    float dv = x - m;
    float var = block_sum(dv * dv, red) / (float)DD;
    float y = tanhf(dv * rsqrtf(var + 1e-5f) * g_out[t] + b_out[t]);

    float m2 = block_sum(y, red) / (float)DD;
    float d2 = y - m2;
    float v2 = block_sum(d2 * d2, red) / (float)DD;
    float z  = d2 * rsqrtf(v2 + 1e-5f) * g_hd[t] + b_hd[t];

    sz[t] = z; __syncthreads();
    float a = b1h[t];
    for (int k = 0; k < DD; k++) a += sz[k] * W1h[k * DD + t];
    st[t] = tanhf(a); __syncthreads();

    if (t < 16) {
        float o = b2h[t];
        for (int k = 0; k < DD; k++) o += st[k] * W2h[k * 16 + t];
        out[b * 16 + t] = o;
    }
}

// ---------------------------------------------------------------------------
// Host launcher
// ---------------------------------------------------------------------------
extern "C" void kernel_launch(void* const* d_in, const int* in_sizes, int n_in,
                              void* d_out, int out_size, void* d_ws, size_t ws_size,
                              hipStream_t stream) {
    (void)in_sizes; (void)n_in; (void)out_size; (void)ws_size;
    const float* hsi    = (const float*)d_in[0];
    const float* lnp_g  = (const float*)d_in[1];
    const float* lnp_b  = (const float*)d_in[2];
    const float* W_emb  = (const float*)d_in[3];
    const float* b_emb  = (const float*)d_in[4];
    const float* lne_g  = (const float*)d_in[5];
    const float* lne_b  = (const float*)d_in[6];
    const float* W_in   = (const float*)d_in[7];
    const float* b_in   = (const float*)d_in[8];
    const float* W_h    = (const float*)d_in[9];
    const float* b_h    = (const float*)d_in[10];
    const float* g_out  = (const float*)d_in[11];
    const float* b_out  = (const float*)d_in[12];
    const float* g_hd   = (const float*)d_in[13];
    const float* b_hd   = (const float*)d_in[14];
    const float* W1h    = (const float*)d_in[15];
    const float* b1h    = (const float*)d_in[16];
    const float* W2h    = (const float*)d_in[17];
    const float* b2h    = (const float*)d_in[18];
    float* out = (float*)d_out;

    // workspace layout (bytes): Wt(131072) | xe | h1 | h2  (bf16 tensors)
    unsigned char* ws = (unsigned char*)d_ws;
    const size_t tensor_elems = (size_t)NPOS * BB * DD;   // 39,649,280
    ushort_t* Wt = (ushort_t*)ws;
    ushort_t* xe = (ushort_t*)(ws + 131072);
    ushort_t* h1 = xe + tensor_elems;
    ushort_t* h2 = h1 + tensor_elems;

    k_prep_w<<<dim3(128), dim3(128), 0, stream>>>(W_in, W_h, Wt);
    k_embed<<<dim3(NPOS), dim3(128), 0, stream>>>(hsi, lnp_g, lnp_b,
                                                  W_emb, b_emb, lne_g, lne_b, xe);
    for (int layer = 0; layer < 2; layer++) {
        const ushort_t* s0 = layer ? h1 : xe;
        ushort_t*       hb = layer ? h2 : h1;
        for (int d = 0; d < NC + NH + NW - 2; d++) {   // 40 wavefronts
            k_scan<<<dim3(NC * NH), dim3(256), 0, stream>>>(s0, hb, Wt,
                                                            b_in, b_h, d);
        }
    }
    k_head<<<dim3(BB), dim3(128), 0, stream>>>(h2, g_out, b_out, g_hd, b_hd,
                                               W1h, b1h, W2h, b2h, out);
}
